// DeepSetMLP_1984274891475
// MI455X (gfx1250) — compile-verified, matrix operand order checked by reading, not run on hardware
//
#include <hip/hip_runtime.h>
#include <hip/hip_bf16.h>

// Problem constants (match reference)
#define BB   4096
#define NN   512
#define CC   512
#define PHI  128
#define H1   512
#define H2   256

#define WAVES 4   // waves per block in fused kernel; each wave owns a 16-row tile

typedef __attribute__((ext_vector_type(16))) _Float16 v16h;
typedef __attribute__((ext_vector_type(8)))  float    v8f;

// Load a 16x32 f16 WMMA fragment (A-layout, also used for B via transposed
// weights): two 16-byte chunks per lane.
__device__ __forceinline__ v16h load_frag(const _Float16* p0, const _Float16* p1) {
    union { v16h v; int4 q[2]; } f;
    f.q[0] = *(const int4*)p0;
    f.q[1] = *(const int4*)p1;
    return f.v;
}

// ---------------------------------------------------------------------------
// Weight prep: dst is (Nd, Kd) f16 with K contiguous; src is (Kd, Nd) f32.
// dst[n*Kd + k] = (f16) src[k*Nd + n]
// ---------------------------------------------------------------------------
__global__ __launch_bounds__(256) void transpose_f32_to_f16(
    const float* __restrict__ src, _Float16* __restrict__ dst, int Kd, int Nd) {
    int idx = blockIdx.x * 256 + threadIdx.x;
    if (idx < Kd * Nd) {
        int n = idx / Kd;
        int k = idx - n * Kd;
        dst[idx] = (_Float16)src[(size_t)k * Nd + n];
    }
}

// ---------------------------------------------------------------------------
// Per-row class histogram via LDS atomics: counts[b, c] (f16, exact: <= 512)
// ---------------------------------------------------------------------------
__global__ __launch_bounds__(256) void histogram_kernel(
    const int* __restrict__ x, _Float16* __restrict__ counts) {
    __shared__ unsigned cnt[CC];
    const int t   = threadIdx.x;
    const int row = blockIdx.x;
    cnt[t]       = 0u;
    cnt[t + 256] = 0u;
    __syncthreads();
    const int* xr = x + (size_t)row * NN;
    unsigned a = (unsigned)xr[t]       & (CC - 1);
    unsigned b = (unsigned)xr[t + 256] & (CC - 1);
    atomicAdd(&cnt[a], 1u);
    atomicAdd(&cnt[b], 1u);
    __syncthreads();
    counts[(size_t)row * CC + t]       = (_Float16)(float)cnt[t];
    counts[(size_t)row * CC + t + 256] = (_Float16)(float)cnt[t + 256];
}

// ---------------------------------------------------------------------------
// Fused MLP: each wave processes a 16-row batch tile end-to-end.
//   u  = counts @ WphiT^T + N*b_phi          (16 x 512 x 128)
//   h1 = relu(u @ W1T^T + b1)                (16 x 128 x 512)
//   h2 = relu(h1 @ W2T^T + b2)               (16 x 512 x 256)
//   out = h2 . W3 + b3                       (f32 GEMV + shfl reduction)
// Intermediates live in per-wave LDS (h1: 16KB, u/h2 ping-pong: 8KB).
// ---------------------------------------------------------------------------
__global__ __launch_bounds__(WAVES * 32) void fused_mlp_kernel(
    const _Float16* __restrict__ counts,  // (B, C) f16
    const _Float16* __restrict__ WphiT,   // (PHI, C) f16, K-contiguous
    const _Float16* __restrict__ W1T,     // (H1, PHI)
    const _Float16* __restrict__ W2T,     // (H2, H1)
    const float* __restrict__ b_phi, const float* __restrict__ b1,
    const float* __restrict__ b2,
    const float* __restrict__ W3, const float* __restrict__ b3,
    float* __restrict__ out) {
    extern __shared__ _Float16 smem[];
    const int lane = threadIdx.x & 31;
    const int wave = threadIdx.x >> 5;
    const int tile = blockIdx.x * WAVES + wave;
    const int row0 = tile * 16;

    _Float16* h1s = smem + wave * (16 * H1 + 16 * H2);  // 16 x 512
    _Float16* ubs = h1s + 16 * H1;                      // u: 16x128 then h2: 16x256

    const int m    = lane & 15;   // A row / B col / D col index
    const int hi   = lane >> 4;   // K-chunk select, D row offset (+8)
    const int coff = hi * 8;

    // ---------------- Stage 1: u = counts @ Wphi (K = C = 512) --------------
    {
        const _Float16* arow = counts + (size_t)(row0 + m) * CC;
        for (int nt = 0; nt < PHI / 16; ++nt) {
            const int gn = nt * 16 + m;
            const _Float16* brow = WphiT + (size_t)gn * CC;
            __builtin_prefetch(brow + CC, 0, 0);
            v8f acc = {};
#pragma unroll
            for (int k0 = 0; k0 < CC; k0 += 32) {
                v16h a = load_frag(arow + k0 + coff, arow + k0 + 16 + coff);
                v16h b = load_frag(brow + k0 + coff, brow + k0 + 16 + coff);
                acc = __builtin_amdgcn_wmma_f32_16x16x32_f16(
                    false, a, false, b, (short)0, acc, false, false);
            }
            const float bias = (float)NN * b_phi[gn];
#pragma unroll
            for (int v = 0; v < 8; ++v)
                ubs[(v + 8 * hi) * PHI + gn] = (_Float16)(acc[v] + bias);
        }
    }
    __syncthreads();

    // ---------------- Stage 2: h1 = relu(u @ W1) (K = PHI = 128) ------------
    {
        const _Float16* arow = ubs + m * PHI;
        for (int nt = 0; nt < H1 / 16; ++nt) {
            const int gn = nt * 16 + m;
            const _Float16* brow = W1T + (size_t)gn * PHI;
            __builtin_prefetch(brow + PHI, 0, 0);
            v8f acc = {};
#pragma unroll
            for (int k0 = 0; k0 < PHI; k0 += 32) {
                v16h a = load_frag(arow + k0 + coff, arow + k0 + 16 + coff);
                v16h b = load_frag(brow + k0 + coff, brow + k0 + 16 + coff);
                acc = __builtin_amdgcn_wmma_f32_16x16x32_f16(
                    false, a, false, b, (short)0, acc, false, false);
            }
            const float bias = b1[gn];
#pragma unroll
            for (int v = 0; v < 8; ++v)
                h1s[(v + 8 * hi) * H1 + gn] = (_Float16)fmaxf(acc[v] + bias, 0.0f);
        }
    }
    __syncthreads();

    // ---------------- Stage 3: h2 = relu(h1 @ W2) (K = H1 = 512) ------------
    {
        const _Float16* arow = h1s + m * H1;
        for (int nt = 0; nt < H2 / 16; ++nt) {
            const int gn = nt * 16 + m;
            const _Float16* brow = W2T + (size_t)gn * H1;
            __builtin_prefetch(brow + H1, 0, 0);
            v8f acc = {};
#pragma unroll
            for (int k0 = 0; k0 < H1; k0 += 32) {
                v16h a = load_frag(arow + k0 + coff, arow + k0 + 16 + coff);
                v16h b = load_frag(brow + k0 + coff, brow + k0 + 16 + coff);
                acc = __builtin_amdgcn_wmma_f32_16x16x32_f16(
                    false, a, false, b, (short)0, acc, false, false);
            }
            const float bias = b2[gn];
#pragma unroll
            for (int v = 0; v < 8; ++v)
                ubs[(v + 8 * hi) * H2 + gn] = (_Float16)fmaxf(acc[v] + bias, 0.0f);
        }
    }
    __syncthreads();

    // ---------------- Stage 4: out = h2 . W3 + b3 (f32 GEMV) ----------------
    {
        const _Float16* hrow = ubs + m * H2 + hi * (H2 / 2);
        const float*    wcol = W3 + hi * (H2 / 2);
        float acc = 0.0f;
#pragma unroll 8
        for (int j = 0; j < H2 / 2; ++j)
            acc += (float)hrow[j] * wcol[j];
        acc += __shfl_xor(acc, 16, 32);  // combine the two K-halves
        if (lane < 16) out[row0 + m] = acc + b3[0];
    }
}

// ---------------------------------------------------------------------------
extern "C" void kernel_launch(void* const* d_in, const int* in_sizes, int n_in,
                              void* d_out, int out_size, void* d_ws, size_t ws_size,
                              hipStream_t stream) {
    (void)in_sizes; (void)n_in; (void)out_size; (void)ws_size;

    const int*   x     = (const int*)d_in[0];
    const float* W_phi = (const float*)d_in[1];  // (C, PHI)
    const float* b_phi = (const float*)d_in[2];
    const float* W1    = (const float*)d_in[3];  // (PHI, H1)
    const float* b1    = (const float*)d_in[4];
    const float* W2    = (const float*)d_in[5];  // (H1, H2)
    const float* b2    = (const float*)d_in[6];
    const float* W3    = (const float*)d_in[7];  // (H2, 1)
    const float* b3    = (const float*)d_in[8];
    float*       out   = (float*)d_out;

    // Workspace layout
    char* ws = (char*)d_ws;
    _Float16* counts = (_Float16*)ws;                          // B*C      = 4 MB
    _Float16* WphiT  = (_Float16*)(ws + (size_t)BB * CC * 2);  // PHI*C    = 128 KB
    _Float16* W1T    = WphiT + (size_t)PHI * CC;               // H1*PHI   = 128 KB
    _Float16* W2T    = W1T + (size_t)H1 * PHI;                 // H2*H1    = 256 KB

    // Weight prep (tiny)
    transpose_f32_to_f16<<<(PHI * CC + 255) / 256, 256, 0, stream>>>(W_phi, WphiT, CC, PHI);
    transpose_f32_to_f16<<<(H1 * PHI + 255) / 256, 256, 0, stream>>>(W1, W1T, PHI, H1);
    transpose_f32_to_f16<<<(H2 * H1 + 255) / 256, 256, 0, stream>>>(W2, W2T, H1, H2);

    // Per-row histograms (turns the gather-sum into a GEMM)
    histogram_kernel<<<BB, 256, 0, stream>>>(x, counts);

    // Fused WMMA MLP: 4096/16 = 256 wave-tiles, 4 waves/block -> 64 blocks
    const int smem_bytes = WAVES * (16 * H1 + 16 * H2) * (int)sizeof(_Float16);  // 96 KB
    fused_mlp_kernel<<<BB / (16 * WAVES), WAVES * 32, smem_bytes, stream>>>(
        counts, WphiT, W1T, W2T, b_phi, b1, b2, W3, b3, out);
}
